// RNNPytorch_19928648254066
// MI455X (gfx1250) — matrix operand order repaired and checked by reading, not compiled, not run on hardware
//
#include <hip/hip_runtime.h>
#include <hip/hip_bf16.h>
#include <math.h>

// ---------------------------------------------------------------------------
// Types for CDNA5 WMMA (wave32): D(16x16 f32) = A(16x32 bf16) x B(32x16 bf16) + C
// ---------------------------------------------------------------------------
typedef __attribute__((ext_vector_type(16))) __bf16 v16bf;
typedef __attribute__((ext_vector_type(8)))  __bf16 v8bf;
typedef __attribute__((ext_vector_type(8)))  float  v8f;

__device__ __forceinline__ v8f wmma_bf16(v16bf a, v16bf b, v8f c) {
  // (neg_a, A, neg_b, B, c_mod, C, reuse_a, reuse_b)
  return __builtin_amdgcn_wmma_f32_16x16x32_bf16(false, a, false, b, (short)0, c,
                                                 false, false);
}

__device__ __forceinline__ v16bf frag_cat(v8bf lo, v8bf hi) {
  return __builtin_shufflevector(lo, hi, 0, 1, 2, 3, 4, 5, 6, 7,
                                 8, 9, 10, 11, 12, 13, 14, 15);
}

// A-fragment (16x32 bf16, MxK) from row-major LDS [M][stride].
// Layout: lanes 0-15 -> M=lane, K=0..15; lanes 16-31 -> M=lane-16, K=16..31
//         (16 consecutive K per lane, matching load_frag_b)
__device__ __forceinline__ v16bf load_frag_a(const __bf16* s, int stride,
                                             int m0, int k0) {
  const int l = threadIdx.x & 31;
  const int row = m0 + (l & 15);
  const int kb = k0 + ((l >> 4) << 4);          // +16 for upper lane half
  const __bf16* p = s + row * stride + kb;
  v8bf lo = *(const v8bf*)(p);                  // K = kb .. kb+7
  v8bf hi = *(const v8bf*)(p + 8);              // K = kb+8 .. kb+15
  return frag_cat(lo, hi);
}

// B-fragment (32x16 bf16, KxN) from LDS laid out [N][stride] (K contiguous).
// Layout used here (contiguous K, same as A): lane n%16 holds column n; lanes 0-15: K=0..15, lanes 16-31: K=16..31,
// VGPR v holds K = base+2v, base+2v+1 (16 consecutive K per lane).
__device__ __forceinline__ v16bf load_frag_b(const __bf16* s, int stride,
                                             int n0, int k0) {
  const int l = threadIdx.x & 31;
  const int col = n0 + (l & 15);
  const int kb = k0 + ((l >> 4) << 4);          // +16 for upper lane half
  const __bf16* p = s + col * stride + kb;
  v8bf lo = *(const v8bf*)(p);
  v8bf hi = *(const v8bf*)(p + 8);
  return frag_cat(lo, hi);
}

// ---------------------------------------------------------------------------
// Phase A / C: parallel GEMM  out[M,512] = A[M,K](f32) @ W[K,512](f32) + bias
// Block: 256 threads (8 waves). Block tile 128(M) x 128(N); wave tile 32x64.
// REMAP==1: store to x_proj layout [T=2048][B=32][512]; batch index is uniform
// per tile (2048 % 128 == 0) -> branchless base + constant stride stores.
// ---------------------------------------------------------------------------
#define BLK_M 128
#define BLK_N 128
#define KB    32
#define APAD  40   // LDS row stride (elems), bank-conflict padding

template <int REMAP>
__global__ void gemm_bf16_kernel(const float* __restrict__ A,
                                 const float* __restrict__ W,
                                 const float* __restrict__ bias,
                                 float* __restrict__ out,
                                 int lda, int K) {
  __shared__ __align__(16) __bf16 sA[BLK_M * APAD];   // [m][k]
  __shared__ __align__(16) __bf16 sB[BLK_N * APAD];   // [n][k] (transposed)

  const int tid = threadIdx.x;
  const int wave = tid >> 5;
  const int l = tid & 31;
  const int m_block = blockIdx.x * BLK_M;
  const int n_block = blockIdx.y * BLK_N;
  const int wm = (wave & 3) * 32;                     // wave M offset in block
  const int wn = (wave >> 2) * 64;                    // wave N offset in block

  v8f acc[2][4] = {};

  for (int k0 = 0; k0 < K; k0 += KB) {
    __syncthreads();
    // Stage A tile: 128x32 f32 -> bf16 (4 float4 per thread)
#pragma unroll
    for (int i = 0; i < 4; ++i) {
      const int slot = tid + i * 256;                 // 1024 float4 slots
      const int row = slot >> 3;
      const int c4 = (slot & 7) << 2;
      const float4 v =
          *(const float4*)(A + (size_t)(m_block + row) * lda + k0 + c4);
      __bf16* d = sA + row * APAD + c4;
      d[0] = (__bf16)v.x; d[1] = (__bf16)v.y;
      d[2] = (__bf16)v.z; d[3] = (__bf16)v.w;
    }
    // Stage B tile transposed: W[k0+kr][n_block+c] -> sB[c][kr]  (32x128)
#pragma unroll
    for (int i = 0; i < 4; ++i) {
      const int slot = tid + i * 256;                 // 1024 float4 slots
      const int kr = slot >> 5;
      const int c4 = (slot & 31) << 2;
      const float4 v =
          *(const float4*)(W + (size_t)(k0 + kr) * 512 + n_block + c4);
      sB[(c4 + 0) * APAD + kr] = (__bf16)v.x;
      sB[(c4 + 1) * APAD + kr] = (__bf16)v.y;
      sB[(c4 + 2) * APAD + kr] = (__bf16)v.z;
      sB[(c4 + 3) * APAD + kr] = (__bf16)v.w;
    }
    __syncthreads();

    const v16bf a0 = load_frag_a(sA, APAD, wm, 0);
    const v16bf a1 = load_frag_a(sA, APAD, wm + 16, 0);
    v16bf b[4];
#pragma unroll
    for (int j = 0; j < 4; ++j) b[j] = load_frag_b(sB, APAD, wn + j * 16, 0);
#pragma unroll
    for (int j = 0; j < 4; ++j) {
      acc[0][j] = wmma_bf16(a0, b[j], acc[0][j]);
      acc[1][j] = wmma_bf16(a1, b[j], acc[1][j]);
    }
  }

  // C layout: lanes 0-15 -> N=lane, M=vgpr; lanes 16-31 -> N=lane-16, M=vgpr+8
  const int cn = n_block + wn + (l & 15);
  const int rbase = m_block + wm + ((l >> 4) << 3);
  const size_t vstride = REMAP ? (size_t)(32 * 512) : (size_t)512;
#pragma unroll
  for (int mi = 0; mi < 2; ++mi) {
    const int row0 = rbase + mi * 16;
    size_t base;
    if (REMAP) {
      const int bb = row0 >> 11;        // uniform within the tile
      const int tt = row0 & 2047;
      base = ((size_t)tt * 32 + bb) * 512;
    } else {
      base = (size_t)row0 * 512;
    }
#pragma unroll
    for (int j = 0; j < 4; ++j) {
      const int col = cn + j * 16;
      const float bi = bias[col];
      float* op = out + base + col;
#pragma unroll
      for (int v = 0; v < 8; ++v) op[v * vstride] = acc[mi][j][v] + bi;
    }
  }
}

// ---------------------------------------------------------------------------
// Phase B: sequential scan. One persistent 1024-thread workgroup (32 waves) on
// one WGP. Each wave owns a 16-wide N-tile of W_hid. Weights are split to fit
// the per-wave VGPR budget at 8 waves/SIMD:
//   - K-blocks 0..7  (K =   0..255): register-resident B-fragments (64 VGPRs)
//   - K-blocks 8..15 (K = 256..511): preloaded ONCE into LDS in B-fragment
//     layout [n][k] (padded stride 264 -> conflict-free ds_load_b128)
// LDS: 270 KB weights + 33 KB single-buffered h = ~304 KB of the WGP's 320 KB.
// A-fragments are one-stage software-pipelined (kb+1 loads issued before kb's
// WMMAs) so LDS latency overlaps the matrix pipe inside each wave.
// x_proj[t] is loaded at the top of step t (consumed only after the 32-WMMA
// chain); x_proj[t+1] is fully covered by per-lane global_prefetch_b8.
//   h_new = tanh(x_proj[t] + h_prev @ W_hid)   (fp32 accum via WMMA bf16)
// ---------------------------------------------------------------------------
#define HSTRIDE 520    // 512 + 8 pad elems; row = 1040B (16B-aligned)
#define RES_KB  8      // register-resident K-blocks
#define LDS_KB  (16 - RES_KB)
#define WSTRIDE (LDS_KB * 32 + 8)   // 264 elems = 528B rows (16B-aligned, padded)

__global__ void rnn_scan_kernel(const float* __restrict__ h0,
                                const float* __restrict__ W_hid,
                                const float* __restrict__ xp,   // [T][B][512]
                                float* __restrict__ hs) {       // [B][T][512]
  __shared__ __align__(16) __bf16 hbuf[32 * HSTRIDE];
  __shared__ __align__(16) __bf16 wlds[512 * WSTRIDE];  // W_hid K=256..511, [n][k]

  const int tid = threadIdx.x;
  const int wave = tid >> 5;
  const int l = tid & 31;
  const int n = wave * 16 + (l & 15);      // this lane's output column
  const int rm = (l >> 4) << 3;            // C-fragment row base for this lane

  // Register-resident W_hid B-fragments for K-blocks 0..RES_KB-1 (fp32->bf16).
  v16bf bw[RES_KB];
#pragma unroll
  for (int kb = 0; kb < RES_KB; ++kb) {
    const int kbase = kb * 32 + ((l >> 4) << 4);
    v16bf f;
#pragma unroll
    for (int i = 0; i < 16; ++i)
      f[i] = (__bf16)W_hid[(size_t)(kbase + i) * 512 + n];
    bw[kb] = f;
  }

  // LDS-resident W_hid for K-blocks RES_KB..15: wlds[n][k - RES_KB*32].
#pragma unroll
  for (int i = 0; i < (512 * LDS_KB * 32) / 1024; ++i) {
    const int e = tid + i * 1024;          // 131072 elems
    const int k = e >> 9;                  // 0..255
    const int nn = e & 511;
    wlds[nn * WSTRIDE + k] = (__bf16)W_hid[(size_t)(RES_KB * 32 + k) * 512 + nn];
  }

  // h0 -> LDS (bf16)
#pragma unroll
  for (int i = 0; i < 16; ++i) {
    const int e = tid + i * 1024;          // 32*512 elems
    hbuf[(e >> 9) * HSTRIDE + (e & 511)] = (__bf16)h0[e];
  }
  __syncthreads();

  for (int t = 0; t < 2048; ++t) {
    // Load this step's x_proj now; consumed only after the WMMA chain.
    const float* xpt = xp + (size_t)t * (32 * 512);
    float xc0[8], xc1[8];
#pragma unroll
    for (int v = 0; v < 8; ++v) {
      xc0[v] = xpt[(rm + v) * 512 + n];
      xc1[v] = xpt[(rm + v + 16) * 512 + n];
    }
    // Full-coverage prefetch of x_proj[t+1]: lane l covers row l, wave covers
    // its own 16-column chunk (one cacheline per lane) -> global_prefetch_b8.
    {
      const int tp = (t < 2047) ? (t + 1) : 2047;
      __builtin_prefetch(xp + (size_t)tp * (32 * 512) + l * 512 + wave * 16,
                         0, 3);
    }

    v8f acc0 = {}, acc1 = {};
    // One-stage pipelined A-fragments: issue kb+1's loads before kb's WMMAs.
    v16bf a0 = load_frag_a(hbuf, HSTRIDE, 0, 0);
    v16bf a1 = load_frag_a(hbuf, HSTRIDE, 16, 0);
#pragma unroll
    for (int kb = 0; kb < 16; ++kb) {
      v16bf bf;
      if (kb < RES_KB) {
        bf = bw[kb];
      } else {
        bf = load_frag_b(wlds, WSTRIDE, wave * 16, (kb - RES_KB) * 32);
      }
      v16bf a0n = a0, a1n = a1;
      if (kb < 15) {
        a0n = load_frag_a(hbuf, HSTRIDE, 0, (kb + 1) * 32);
        a1n = load_frag_a(hbuf, HSTRIDE, 16, (kb + 1) * 32);
      }
      acc0 = wmma_bf16(a0, bf, acc0);
      acc1 = wmma_bf16(a1, bf, acc1);
      a0 = a0n;
      a1 = a1n;
    }
    __syncthreads();   // all A-fragment reads of h done

#pragma unroll
    for (int v = 0; v < 8; ++v) {
      const int m0 = rm + v;               // batch rows 0..15
      const float hv0 = tanhf(acc0[v] + xc0[v]);
      hs[((size_t)m0 * 2048 + t) * 512 + n] = hv0;
      hbuf[m0 * HSTRIDE + n] = (__bf16)hv0;

      const int m1 = m0 + 16;              // batch rows 16..31
      const float hv1 = tanhf(acc1[v] + xc1[v]);
      hs[((size_t)m1 * 2048 + t) * 512 + n] = hv1;
      hbuf[m1 * HSTRIDE + n] = (__bf16)hv1;
    }
    __syncthreads();   // new h visible to all waves
  }
}

// ---------------------------------------------------------------------------
// Launch: A (x-projection GEMM, x_proj staged into the outs half of d_out in
// [T,B,H] order) -> B (scan, writes hs) -> C (output GEMM, overwrites x_proj
// region with the real outs). Stream ordering provides the dependencies.
// ---------------------------------------------------------------------------
extern "C" void kernel_launch(void* const* d_in, const int* in_sizes, int n_in,
                              void* d_out, int out_size, void* d_ws,
                              size_t ws_size, hipStream_t stream) {
  const float* x     = (const float*)d_in[0];  // [32,2048,256]
  const float* h0    = (const float*)d_in[1];  // [32,512]
  // d_in[2] = c, unused by the reference math
  const float* W_in  = (const float*)d_in[3];  // [256,512]
  const float* W_hid = (const float*)d_in[4];  // [512,512]
  const float* b_hid = (const float*)d_in[5];  // [512]
  const float* W_out = (const float*)d_in[6];  // [512,512]
  const float* b_out = (const float*)d_in[7];  // [512]

  float* hs   = (float*)d_out;                            // [32,2048,512]
  float* reg1 = hs + (size_t)32 * 2048 * 512;             // x_proj, then outs

  const dim3 blk(256);
  const dim3 grd(65536 / BLK_M, 512 / BLK_N);

  // Phase A: x_proj[t][b][:] = x[b][t][:] @ W_in + b_hid   (REMAP=1)
  gemm_bf16_kernel<1><<<grd, blk, 0, stream>>>(x, W_in, b_hid, reg1,
                                               /*lda=*/256, /*K=*/256);
  // Phase B: sequential recurrence, writes hs
  rnn_scan_kernel<<<1, 1024, 0, stream>>>(h0, W_hid, reg1, hs);
  // Phase C: outs[b][t][:] = hs[b][t][:] @ W_out + b_out   (REMAP=0)
  gemm_bf16_kernel<0><<<grd, blk, 0, stream>>>(hs, W_out, b_out, reg1,
                                               /*lda=*/512, /*K=*/512);
}
